// NSACore_5772436046578
// MI455X (gfx1250) — matrix operand order repaired and checked by reading, not compiled, not run on hardware
//
#include <hip/hip_runtime.h>
#include <hip/hip_bf16.h>

typedef __attribute__((ext_vector_type(16))) _Float16 v16h;
typedef __attribute__((ext_vector_type(8)))  float    v8f;
typedef int v4i __attribute__((vector_size(16)));

#define T_Q   16
#define KHN   2
#define GN    8
#define DN    128
#define VDN   128
#define BSN   32
#define NBN   64
#define TOPKN 8
#define WINN  256

static constexpr float SCALE = 0.0883883476483184f; // 1/sqrt(128)

// ---------------- async global->LDS staging (CDNA5 ASYNCcnt path) ----------------
#if __has_builtin(__builtin_amdgcn_global_load_async_to_lds_b128) && \
    __has_builtin(__builtin_amdgcn_s_wait_asynccnt)
#define HAVE_ASYNC 1
#define ASYNC_WAIT(n) __builtin_amdgcn_s_wait_asynccnt(n)
static __device__ __forceinline__ void async_cp16(const _Float16* g, _Float16* l) {
  __builtin_amdgcn_global_load_async_to_lds_b128(
      (__attribute__((address_space(1))) v4i*)g,
      (__attribute__((address_space(3))) v4i*)l, 0, 0);
}
#else
#define HAVE_ASYNC 0
#define ASYNC_WAIT(n)
#endif

// stage a 32x128 f16 row-major chunk (rows = tokens/blocks, src row stride src_ld elems)
static __device__ __forceinline__ void stage_k16(_Float16* dst, const _Float16* src,
                                                 int src_ld, int lane) {
#if HAVE_ASYNC
#pragma unroll
  for (int i = 0; i < 16; ++i) {            // 512 x 16B segments / 32 lanes
    int idx = lane + 32 * i;
    int r = idx >> 4, s = (idx & 15) * 8;
    async_cp16(src + r * src_ld + s, dst + r * DN + s);
  }
#else
  for (int idx = lane; idx < BSN * DN; idx += 32) {
    int r = idx >> 7, d = idx & 127;
    dst[idx] = src[r * src_ld + d];
  }
#endif
}
// stage a transposed value chunk: dst[d*32 + tok] from pre-transposed src[d*src_ld + tok]
static __device__ __forceinline__ void stage_vT16(_Float16* dst, const _Float16* src,
                                                  int src_ld, int lane) {
#if HAVE_ASYNC
#pragma unroll
  for (int i = 0; i < 16; ++i) {            // 128 rows x 64B = 512 x 16B segments
    int idx = lane + 32 * i;
    int d = idx >> 2, s = (idx & 3) * 8;
    async_cp16(src + d * src_ld + s, dst + d * 32 + s);
  }
#else
  for (int idx = lane; idx < BSN * VDN; idx += 32) {
    int r = idx & 31, d = idx >> 5;
    dst[d * 32 + r] = src[d * src_ld + r];
  }
#endif
}

// ---- WMMA fragment loaders (layouts per CDNA5 ISA §7.12.2) ----
static __device__ __forceinline__ v16h frag_A(const _Float16* base, int ld, int row,
                                              int kbase, int half) {
  v16h a;
#pragma unroll
  for (int e = 0; e < 16; ++e) {
    int kk = (e < 8) ? (half * 8 + e) : (16 + half * 8 + (e - 8));
    a[e] = base[row * ld + kbase + kk];
  }
  return a;
}
// B[k][n] = mat[n][k] (mat row-major, row stride ld): two 16B runs per fragment
static __device__ __forceinline__ v16h frag_B_rows(const _Float16* mat, int ld, int col,
                                                   int kbase, int half) {
  v16h b;
#pragma unroll
  for (int e = 0; e < 16; ++e)
    b[e] = mat[col * ld + kbase + ((e < 8) ? (half * 8 + e) : (16 + half * 8 + (e - 8)))];
  return b;
}

// S[rows][colbase..colbase+31] = lq(16x128) x staged-keys(32x128)^T, f32 accum.
template <bool FULL>
static __device__ __forceinline__ void score_chunk(const _Float16* lq, const _Float16* kch,
                                                   float* sbuf, int srow, int colbase,
                                                   int lane) {
  const int half = lane >> 4, lrow = lane & 15;
  v8f acc0 = {}, acc1 = {};
#pragma unroll
  for (int kk = 0; kk < 4; ++kk) {
    v16h a  = frag_A(lq, DN, lrow, kk * 32, half);
    v16h b0 = frag_B_rows(kch, DN, lrow,      kk * 32, half);
    v16h b1 = frag_B_rows(kch, DN, 16 + lrow, kk * 32, half);
    acc0 = __builtin_amdgcn_wmma_f32_16x16x32_f16(false, a, false, b0, (short)0, acc0,
                                                  false, false);
    acc1 = __builtin_amdgcn_wmma_f32_16x16x32_f16(false, a, false, b1, (short)0, acc1,
                                                  false, false);
  }
  if (FULL || half == 0) {
#pragma unroll
    for (int r = 0; r < 8; ++r) {
      int m = r + 8 * half;
      sbuf[m * srow + colbase + lrow]      = acc0[r];
      sbuf[m * srow + colbase + 16 + lrow] = acc1[r];
    }
  }
}

// ---------------- prep: k -> f16, v -> f16 transposed [kh][d][t] ----------------
__global__ __launch_bounds__(256) void nsa_prep_kernel(
    const float* __restrict__ k, const float* __restrict__ v,
    _Float16* __restrict__ k16, _Float16* __restrict__ vT16, int T) {
  int i = blockIdx.x * 256 + threadIdx.x;
  if (i < T * KHN * DN) {
    k16[i] = (_Float16)k[i];
    int d = i & 127, kh = (i >> 7) & 1, t = i >> 8;
    vT16[(kh * VDN + d) * T + t] = (_Float16)v[i];
  }
}

// ------- compression: swiglu(blk @ W^T); ck16 row-major, cvT16 transposed -------
__global__ __launch_bounds__(256) void nsa_compress_kernel(
    const float* __restrict__ k, const float* __restrict__ v,
    const float* __restrict__ wk, const float* __restrict__ wv,
    _Float16* __restrict__ ck16, _Float16* __restrict__ cvT16) {
  const int n = blockIdx.x, h = blockIdx.y, which = blockIdx.z;
  const float* x = which ? v : k;
  const float* w = which ? wv : wk;
  __shared__ float xblk[BSN * DN];
  __shared__ float hbuf[2 * DN];
  const int tid = threadIdx.x;
  for (int j = tid; j < BSN * DN; j += 256) {
    int i = j >> 7, d = j & 127;
    xblk[j] = x[((n * BSN + i) * KHN + h) * DN + d];
  }
  __syncthreads();
  const float* wr = w + tid * (BSN * DN);
  float acc = 0.f;
  for (int j = 0; j < BSN * DN; ++j) acc += wr[j] * xblk[j];
  hbuf[tid] = acc;
  __syncthreads();
  if (tid < DN) {
    float a = hbuf[tid], b = hbuf[tid + DN];
    float o = a * (b / (1.f + __expf(-b)));
    if (which == 0) ck16[(n * KHN + h) * DN + tid] = (_Float16)o;
    else            cvT16[(h * VDN + tid) * NBN + n] = (_Float16)o;
  }
}

// -------- compressed-KV attention + top-k selection (2 queries / wave) --------
__global__ __launch_bounds__(32) void nsa_cmp_kernel(
    const float* __restrict__ q, const float* __restrict__ cw,
    const _Float16* __restrict__ ck16, const _Float16* __restrict__ cvT16,
    int* __restrict__ sel_out, float* __restrict__ out) {
  const int t0 = blockIdx.x * 2, kh = blockIdx.y, lane = threadIdx.x;
  const int half = lane >> 4, lrow = lane & 15;
  __shared__ _Float16 lq[16 * DN];
  __shared__ _Float16 chbufA[BSN * DN];
  __shared__ _Float16 chbufB[BSN * DN];
  __shared__ float    sbuf[16 * NBN];
  __shared__ _Float16 pbuf[16 * NBN];
  __shared__ float    impL[2][NBN];

  for (int idx = lane; idx < 16 * DN; idx += 32) {  // rows 0..7 = t0, 8..15 = t0+1
    int r = idx >> 7, d = idx & 127;
    int tq = t0 + (r >> 3), g = r & 7;
    lq[idx] = (_Float16)(q[(tq * T_Q + kh * GN + g) * DN + d] * SCALE);
  }
  __syncthreads();

  const int nvalid0 = (t0 + 1) >> 5, nvalid1 = (t0 + 2) >> 5;
  const int nchunk  = (nvalid1 + 31) >> 5;                      // 0..2

  if (nchunk > 0) stage_k16(chbufA, ck16 + kh * DN, KHN * DN, lane);
  for (int c = 0; c < nchunk; ++c) {
    const _Float16* cur = (c & 1) ? chbufB : chbufA;
    if (c + 1 < nchunk) {
      stage_k16((c & 1) ? chbufA : chbufB, ck16 + ((c + 1) * 32 * KHN + kh) * DN,
                KHN * DN, lane);
      ASYNC_WAIT(16);
    } else ASYNC_WAIT(0);
    __syncthreads();
    score_chunk<true>(lq, cur, sbuf, NBN, c * 32, lane);
    __syncthreads();
  }

  // overlap P@V operand DMA with softmax + selection
  stage_vT16(chbufA, cvT16 + kh * VDN * NBN,      NBN, lane);
  stage_vT16(chbufB, cvT16 + kh * VDN * NBN + 32, NBN, lane);

  {  // masked softmax: 16 rows, 2 lanes per row
    int row = lane >> 1, sub = lane & 1;
    int nv = (row < 8) ? nvalid0 : nvalid1;
    float m = -1e30f;
    for (int c = sub; c < nv; c += 2) m = fmaxf(m, sbuf[row * NBN + c]);
    m = fmaxf(m, __shfl_xor(m, 1, 2));
    float s = 0.f;
    for (int c = sub; c < NBN; c += 2) {
      float p = (c < nv) ? __expf(sbuf[row * NBN + c] - m) : 0.f;
      sbuf[row * NBN + c] = p; s += p;
    }
    s += __shfl_xor(s, 1, 2);
    float inv = 1.f / fmaxf(s, 1e-20f);
    for (int c = sub; c < NBN; c += 2) {
      float p = sbuf[row * NBN + c] * inv;
      sbuf[row * NBN + c] = p;
      pbuf[row * NBN + c] = (_Float16)p;
    }
  }
  __syncthreads();

  for (int n = lane; n < NBN; n += 32) {  // importance per query (group-summed probs)
    float v0 = -1.f, v1 = -1.f;
    if (n < nvalid0) { v0 = 0.f; for (int g = 0; g < 8;  ++g) v0 += sbuf[g * NBN + n]; }
    if (n < nvalid1) { v1 = 0.f; for (int g = 8; g < 16; ++g) v1 += sbuf[g * NBN + n]; }
    impL[0][n] = v0; impL[1][n] = v1;
  }
  __syncthreads();
  if (lane < 2) {                          // stable top-k (ties -> lowest index)
    unsigned long long used = 0ull;
    for (int j = 0; j < TOPKN; ++j) {
      float bv = -1e30f; int bi = 0;
      for (int n = 0; n < NBN; ++n) {
        if ((used >> n) & 1ull) continue;
        if (impL[lane][n] > bv) { bv = impL[lane][n]; bi = n; }
      }
      used |= 1ull << bi;
      sel_out[((t0 + lane) * KHN + kh) * TOPKN + j] = bi;
    }
  }

  v8f acc[8] = {};                         // O = P @ CV
  for (int c = 0; c < 2; ++c) {
    const _Float16* cur = (c & 1) ? chbufB : chbufA;
    if (c == 0) ASYNC_WAIT(16); else ASYNC_WAIT(0);
    __syncthreads();
    v16h a = frag_A(pbuf, NBN, lrow, c * 32, half);
#pragma unroll
    for (int nt = 0; nt < 8; ++nt) {
      v16h b = frag_B_rows(cur, 32, nt * 16 + lrow, 0, half);
      acc[nt] = __builtin_amdgcn_wmma_f32_16x16x32_f16(false, a, false, b, (short)0,
                                                       acc[nt], false, false);
    }
    __syncthreads();
  }

  const int tq = t0 + half;                // all 32 lanes write (row M = r + 8*half)
#pragma unroll
  for (int r = 0; r < 8; ++r) {
    int qh = kh * GN + r;
    float gate = cw[(tq * T_Q + qh) * 3 + 0];
#pragma unroll
    for (int nt = 0; nt < 8; ++nt)
      out[(tq * T_Q + qh) * VDN + nt * 16 + lrow] = gate * acc[nt][r];
  }
}

// ---------------- selected-block attention (1 query / wave) ----------------
__global__ __launch_bounds__(32) void nsa_slc_kernel(
    const float* __restrict__ q, const _Float16* __restrict__ k16,
    const _Float16* __restrict__ vT16, const float* __restrict__ cw,
    const int* __restrict__ sel_in, float* __restrict__ out, int T) {
  const int t = blockIdx.x, kh = blockIdx.y, lane = threadIdx.x;
  const int half = lane >> 4, lrow = lane & 15;
  __shared__ _Float16 lq[16 * DN];
  __shared__ _Float16 chbufA[BSN * DN];
  __shared__ _Float16 chbufB[BSN * DN];
  __shared__ float    sbuf[GN * 256];
  __shared__ _Float16 pbuf[16 * 256];
  __shared__ int      selL[TOPKN];

  if (lane < TOPKN) selL[lane] = sel_in[(t * KHN + kh) * TOPKN + lane];
  for (int idx = lane; idx < 16 * DN; idx += 32) {
    int r = idx >> 7, d = idx & 127;
    lq[idx] = (r < GN) ? (_Float16)(q[(t * T_Q + kh * GN + r) * DN + d] * SCALE)
                       : (_Float16)0.f;
  }
  for (int idx = 8 * 256 + lane; idx < 16 * 256; idx += 32) pbuf[idx] = (_Float16)0.f;
  __syncthreads();

  stage_k16(chbufA, k16 + (selL[0] * BSN * KHN + kh) * DN, KHN * DN, lane);
  for (int b = 0; b < TOPKN; ++b) {
    const _Float16* cur = (b & 1) ? chbufB : chbufA;
    if (b + 1 < TOPKN) {
      stage_k16((b & 1) ? chbufA : chbufB,
                k16 + (selL[b + 1] * BSN * KHN + kh) * DN, KHN * DN, lane);
      ASYNC_WAIT(16);
    } else ASYNC_WAIT(0);
    __syncthreads();
    score_chunk<false>(lq, cur, sbuf, 256, b * 32, lane);
    __syncthreads();
  }

  stage_vT16(chbufA, vT16 + kh * VDN * T + selL[0] * BSN, T, lane);  // overlap w/ softmax

  {  // masked softmax, causal per gathered token
    int row = lane >> 2, sub = lane & 3;
    float m = -1e30f;
    for (int c = sub; c < 256; c += 4) {
      int tok = selL[c >> 5] * BSN + (c & 31);
      if (tok <= t) m = fmaxf(m, sbuf[row * 256 + c]);
    }
    m = fmaxf(m, __shfl_xor(m, 1, 4));
    m = fmaxf(m, __shfl_xor(m, 2, 4));
    float s = 0.f;
    for (int c = sub; c < 256; c += 4) {
      int tok = selL[c >> 5] * BSN + (c & 31);
      float p = (tok <= t) ? __expf(sbuf[row * 256 + c] - m) : 0.f;
      sbuf[row * 256 + c] = p; s += p;
    }
    s += __shfl_xor(s, 1, 4); s += __shfl_xor(s, 2, 4);
    float inv = 1.f / fmaxf(s, 1e-20f);
    for (int c = sub; c < 256; c += 4)
      pbuf[row * 256 + c] = (_Float16)(sbuf[row * 256 + c] * inv);
  }
  __syncthreads();

  v8f acc[8] = {};
  for (int b = 0; b < TOPKN; ++b) {
    const _Float16* cur = (b & 1) ? chbufB : chbufA;
    if (b + 1 < TOPKN) {
      stage_vT16((b & 1) ? chbufA : chbufB,
                 vT16 + kh * VDN * T + selL[b + 1] * BSN, T, lane);
      ASYNC_WAIT(16);
    } else ASYNC_WAIT(0);
    __syncthreads();
    v16h a = frag_A(pbuf, 256, lrow, b * 32, half);
#pragma unroll
    for (int nt = 0; nt < 8; ++nt) {
      v16h bb = frag_B_rows(cur, 32, nt * 16 + lrow, 0, half);
      acc[nt] = __builtin_amdgcn_wmma_f32_16x16x32_f16(false, a, false, bb, (short)0,
                                                       acc[nt], false, false);
    }
    __syncthreads();
  }

  if (half == 0) {
#pragma unroll
    for (int r = 0; r < 8; ++r) {
      int qh = kh * GN + r;
      float gate = cw[(t * T_Q + qh) * 3 + 1];
#pragma unroll
      for (int nt = 0; nt < 8; ++nt)
        out[(t * T_Q + qh) * VDN + nt * 16 + lrow] += gate * acc[nt][r];
    }
  }
}

// ---------------- sliding-window attention (2 queries / wave) ----------------
__global__ __launch_bounds__(32) void nsa_swa_kernel(
    const float* __restrict__ q, const _Float16* __restrict__ k16,
    const _Float16* __restrict__ vT16, const float* __restrict__ cw,
    float* __restrict__ out, int T) {
  const int t0 = blockIdx.x * 2, t1 = t0 + 1, kh = blockIdx.y, lane = threadIdx.x;
  const int half = lane >> 4, lrow = lane & 15;
  __shared__ _Float16 lq[16 * DN];
  __shared__ _Float16 chbufA[BSN * DN];
  __shared__ _Float16 chbufB[BSN * DN];
  __shared__ float    sbuf[16 * 288];
  __shared__ _Float16 pbuf[16 * 288];

  const int j0  = (t0 >= WINN) ? (t0 - WINN + 1) : 0;
  const int c0  = j0 >> 5, c1 = t1 >> 5;
  const int nch = c1 - c0 + 1;                         // <= 9
  const int ncol = nch * 32;

  for (int idx = lane; idx < 16 * DN; idx += 32) {     // rows 0..7 = t0, 8..15 = t1
    int r = idx >> 7, d = idx & 127;
    int tq = t0 + (r >> 3), g = r & 7;
    lq[idx] = (_Float16)(q[(tq * T_Q + kh * GN + g) * DN + d] * SCALE);
  }
  __syncthreads();

  stage_k16(chbufA, k16 + (c0 * 32 * KHN + kh) * DN, KHN * DN, lane);
  for (int ci = 0; ci < nch; ++ci) {
    const _Float16* cur = (ci & 1) ? chbufB : chbufA;
    if (ci + 1 < nch) {
      stage_k16((ci & 1) ? chbufA : chbufB,
                k16 + ((c0 + ci + 1) * 32 * KHN + kh) * DN, KHN * DN, lane);
      ASYNC_WAIT(16);
    } else ASYNC_WAIT(0);
    __syncthreads();
    score_chunk<true>(lq, cur, sbuf, 288, ci * 32, lane);
    __syncthreads();
  }

  stage_vT16(chbufA, vT16 + kh * VDN * T + c0 * 32, T, lane);  // overlap w/ softmax

  {  // masked softmax: per-row window [tq-255, tq]
    int row = lane >> 1, sub = lane & 1;
    int tq  = t0 + (row >> 3);
    int jlo = (tq >= WINN) ? (tq - WINN + 1) : 0;
    float m = -1e30f;
    for (int c = sub; c < ncol; c += 2) {
      int j = c0 * 32 + c;
      if (j >= jlo && j <= tq) m = fmaxf(m, sbuf[row * 288 + c]);
    }
    m = fmaxf(m, __shfl_xor(m, 1, 2));
    float s = 0.f;
    for (int c = sub; c < ncol; c += 2) {
      int j = c0 * 32 + c;
      float p = (j >= jlo && j <= tq) ? __expf(sbuf[row * 288 + c] - m) : 0.f;
      sbuf[row * 288 + c] = p; s += p;
    }
    s += __shfl_xor(s, 1, 2);
    float inv = 1.f / fmaxf(s, 1e-20f);
    for (int c = sub; c < ncol; c += 2)
      pbuf[row * 288 + c] = (_Float16)(sbuf[row * 288 + c] * inv);
  }
  __syncthreads();

  v8f acc[8] = {};
  for (int ci = 0; ci < nch; ++ci) {
    const _Float16* cur = (ci & 1) ? chbufB : chbufA;
    if (ci + 1 < nch) {
      stage_vT16((ci & 1) ? chbufA : chbufB,
                 vT16 + kh * VDN * T + (c0 + ci + 1) * 32, T, lane);
      ASYNC_WAIT(16);
    } else ASYNC_WAIT(0);
    __syncthreads();
    v16h a = frag_A(pbuf, 288, lrow, ci * 32, half);
#pragma unroll
    for (int nt = 0; nt < 8; ++nt) {
      v16h bb = frag_B_rows(cur, 32, nt * 16 + lrow, 0, half);
      acc[nt] = __builtin_amdgcn_wmma_f32_16x16x32_f16(false, a, false, bb, (short)0,
                                                       acc[nt], false, false);
    }
    __syncthreads();
  }

  const int tq = t0 + half;
#pragma unroll
  for (int r = 0; r < 8; ++r) {
    int qh = kh * GN + r;
    float gate = cw[(tq * T_Q + qh) * 3 + 2];
#pragma unroll
    for (int nt = 0; nt < 8; ++nt)
      out[(tq * T_Q + qh) * VDN + nt * 16 + lrow] += gate * acc[nt][r];
  }
}

extern "C" void kernel_launch(void* const* d_in, const int* in_sizes, int n_in,
                              void* d_out, int out_size, void* d_ws, size_t ws_size,
                              hipStream_t stream) {
  const float* q  = (const float*)d_in[0];
  const float* k  = (const float*)d_in[1];
  const float* v  = (const float*)d_in[2];
  const float* cw = (const float*)d_in[3];
  const float* wk = (const float*)d_in[4];
  const float* wv = (const float*)d_in[5];
  float* out = (float*)d_out;

  const int T = in_sizes[0] / (T_Q * DN);   // 2048 for this harness

  // ws layout: ck16 | cvT16 | k16 | vT16 (f16) then sel (int) : ~2.3 MB total
  _Float16* ck16  = (_Float16*)d_ws;
  _Float16* cvT16 = ck16  + NBN * KHN * DN;
  _Float16* k16   = cvT16 + KHN * VDN * NBN;
  _Float16* vT16  = k16   + (size_t)T * KHN * DN;
  int*      sel   = (int*)(vT16 + (size_t)T * KHN * VDN);

  nsa_prep_kernel<<<dim3((T * KHN * DN + 255) / 256), 256, 0, stream>>>(k, v, k16, vT16, T);
  nsa_compress_kernel<<<dim3(T / BSN, KHN, 2), 256, 0, stream>>>(k, v, wk, wv, ck16, cvT16);
  nsa_cmp_kernel<<<dim3(T / 2, KHN), 32, 0, stream>>>(q, cw, ck16, cvT16, sel, out);
  nsa_slc_kernel<<<dim3(T, KHN), 32, 0, stream>>>(q, k16, vT16, cw, sel, out, T);
  nsa_swa_kernel<<<dim3(T / 2, KHN), 32, 0, stream>>>(q, k16, vT16, cw, out, T);
}
